// Model_44169443672858
// MI455X (gfx1250) — compile-verified, matrix operand order checked by reading, not run on hardware
//
#include <hip/hip_runtime.h>
#include <math.h>

// ---------------------------------------------------------------------------
// Types for CDNA5 WMMA (wave32)
// ---------------------------------------------------------------------------
typedef __attribute__((ext_vector_type(16))) __bf16 v16bf;
typedef __attribute__((ext_vector_type(8)))  __bf16 v8bf;
typedef __attribute__((ext_vector_type(4)))  __bf16 v4bf;
typedef __attribute__((ext_vector_type(8)))  float  v8f;
typedef __attribute__((ext_vector_type(4)))  float  v4f;

#define DEVINL __device__ __forceinline__

DEVINL __bf16 f2bf(float f) {
  unsigned u = __builtin_bit_cast(unsigned, f);
  unsigned r = u + 0x7FFFu + ((u >> 16) & 1u);   // round-to-nearest-even
  unsigned short h = (unsigned short)(r >> 16);
  return __builtin_bit_cast(__bf16, h);
}

DEVINL float gelu_exact(float x) {
  return 0.5f * x * (1.0f + erff(x * 0.7071067811865476f));
}

// Fragment load: two contiguous 16B runs per lane -> 2x ds_load_b128.
// rowBase must be 16B aligned; h = lane>>4.
DEVINL v16bf ldfrag(const __bf16* rowBase, int h) {
  const v8bf x0 = *(const v8bf*)(rowBase + h * 8);
  const v8bf x1 = *(const v8bf*)(rowBase + 16 + h * 8);
  return __builtin_shufflevector(x0, x1, 0, 1, 2, 3, 4, 5, 6, 7,
                                 8, 9, 10, 11, 12, 13, 14, 15);
}

// Async global->LDS copy, 8 bytes per lane, tracked by ASYNCcnt.
// LDS offset = low 32 bits of the flat shared-aperture address (ISA 10.2).
DEVINL void async_copy_b64(const __bf16* gsrc, __bf16* ldst) {
  unsigned lds_off = (unsigned)(unsigned long long)ldst;
  unsigned long long ga = (unsigned long long)gsrc;
  asm volatile("global_load_async_to_lds_b64 %0, %1, off"
               :: "v"(lds_off), "v"(ga) : "memory");
}

// ---------------------------------------------------------------------------
// Model dimensions
// ---------------------------------------------------------------------------
#define BN   32
#define LSEQ 512
#define DMOD 512
#define DFF  2048
#define CIN  21
#define CMK  4
#define COUT 21
#define MAK  25
#define TOPK 6
#define PRED 256

// ---------------------------------------------------------------------------
// Kernel 1a: fp32 -> bf16 (activations, same layout)
// ---------------------------------------------------------------------------
__global__ void k_f32_to_bf16(const float* __restrict__ in, __bf16* __restrict__ out, long n) {
  long i = (long)blockIdx.x * blockDim.x + threadIdx.x;
  if (i < n) out[i] = f2bf(in[i]);
}

// Kernel 1b: fp32 [K,N] -> bf16 transposed [N,K]  (one-time weight prep)
__global__ void k_transpose_bf16(const float* __restrict__ src, __bf16* __restrict__ dst,
                                 int K, int N) {
  long gid = (long)blockIdx.x * blockDim.x + threadIdx.x;
  if (gid >= (long)K * N) return;
  int k = (int)(gid % K);
  int n = (int)(gid / K);
  dst[gid] = f2bf(src[(long)k * N + n]);   // dst[n*K + k]
}

// ---------------------------------------------------------------------------
// Kernel 2: WMMA bf16 GEMM   C[M,N] = act( A[M,K] * W[K,N] )
// A given as bf16 [M,K]; W as bf16 transposed Wt[N,K].
// Block tile 64x128, K-step 32.  8 waves, each owns a 32x32 strip:
// 2 A-frags x 2 B-frags -> 4 v_wmma per K-step.
// Double-buffered LDS: tile t+1 staged via global_load_async_to_lds_b64
// while tile t computes; wait s_wait_asynccnt<=6 (in-order completion per
// wave, 6 async ops per tile) hides the DMA under the WMMAs.
// Requires: M % 64 == 0, N % 128 == 0, K % 32 == 0 (true for all call sites).
// ---------------------------------------------------------------------------
#define TM   64
#define TN   128
#define TKK  32
#define LDST 40   // LDS row stride in elements (80B: keeps 16B alignment)

__global__ __launch_bounds__(256) void k_gemm_bf16(
    const __bf16* __restrict__ A, const __bf16* __restrict__ Wt,
    float* __restrict__ C, int M, int N, int K, int act)
{
  __shared__ alignas(16) __bf16 lA[2][TM * LDST];
  __shared__ alignas(16) __bf16 lB[2][TN * LDST];

  const int tid  = threadIdx.x;
  const int lane = tid & 31;
  const int wv   = tid >> 5;
  const int wm   = (wv & 1) * 32;   // wave M offset (2 waves along M)
  const int wn   = (wv >> 1) * 32;  // wave N offset (4 waves along N)
  const int half = lane >> 4;
  const int lr   = lane & 15;

  const int blockRow = blockIdx.y * TM;
  const int blockCol = blockIdx.x * TN;

  v8f acc00 = {}, acc01 = {}, acc10 = {}, acc11 = {};

  // stage tile (6 async b64 copies per thread: 2 for A, 4 for B)
  auto stage = [&](int buf, int k0) {
#pragma unroll
    for (int j = 0; j < 2; ++j) {
      int chunk = tid + j * 256;        // 0..511 (64 rows x 8 chunks)
      int r  = chunk >> 3;
      int kc = (chunk & 7) * 4;
      async_copy_b64(&A[(size_t)(blockRow + r) * K + k0 + kc],
                     &lA[buf][r * LDST + kc]);
    }
#pragma unroll
    for (int j = 0; j < 4; ++j) {
      int chunk = tid + j * 256;        // 0..1023 (128 rows x 8 chunks)
      int r  = chunk >> 3;
      int kc = (chunk & 7) * 4;
      async_copy_b64(&Wt[(size_t)(blockCol + r) * K + k0 + kc],
                     &lB[buf][r * LDST + kc]);
    }
  };

  const int nk = K / TKK;
  stage(0, 0);                           // prologue

  for (int t = 0; t < nk; ++t) {
    const int  cur     = t & 1;
    const bool hasNext = (t + 1) < nk;
    if (hasNext) {
      stage(cur ^ 1, (t + 1) * TKK);     // DMA next tile under this tile's math
      asm volatile("s_wait_asynccnt 0x6" ::: "memory");   // tile t landed
    } else {
      asm volatile("s_wait_asynccnt 0x0" ::: "memory");
    }
    __syncthreads();

    v16bf aF0 = ldfrag(&lA[cur][(wm + lr) * LDST], half);
    v16bf aF1 = ldfrag(&lA[cur][(wm + 16 + lr) * LDST], half);
    v16bf bF0 = ldfrag(&lB[cur][(wn + lr) * LDST], half);
    v16bf bF1 = ldfrag(&lB[cur][(wn + 16 + lr) * LDST], half);

    acc00 = __builtin_amdgcn_wmma_f32_16x16x32_bf16(false, aF0, false, bF0,
                                                    (short)0, acc00, false, false);
    acc01 = __builtin_amdgcn_wmma_f32_16x16x32_bf16(false, aF0, false, bF1,
                                                    (short)0, acc01, false, false);
    acc10 = __builtin_amdgcn_wmma_f32_16x16x32_bf16(false, aF1, false, bF0,
                                                    (short)0, acc10, false, false);
    acc11 = __builtin_amdgcn_wmma_f32_16x16x32_bf16(false, aF1, false, bF1,
                                                    (short)0, acc11, false, false);
    __syncthreads();   // done reading buf 'cur' before it is re-staged at t+2
  }

  // D layout: VGPR r -> M = r + 8*half, N = lane%16
#pragma unroll
  for (int r = 0; r < 8; ++r) {
    int m0 = blockRow + wm + half * 8 + r;
    int m1 = m0 + 16;
    int n0 = blockCol + wn + lr;
    int n1 = n0 + 16;
    float o00 = acc00[r], o01 = acc01[r], o10 = acc10[r], o11 = acc11[r];
    if (act == 1) {
      o00 = gelu_exact(o00); o01 = gelu_exact(o01);
      o10 = gelu_exact(o10); o11 = gelu_exact(o11);
    }
    if (m0 < M) {
      C[(size_t)m0 * N + n0] = o00;
      C[(size_t)m0 * N + n1] = o01;
    }
    if (m1 < M) {
      C[(size_t)m1 * N + n0] = o10;
      C[(size_t)m1 * N + n1] = o11;
    }
  }
}

// ---------------------------------------------------------------------------
// Kernel 3: series decomposition.  trend = movmean(x, 25) (edge replicate),
// seasonal = x - trend.  mode 0: discard trend, 1: taccum += trend,
// 2: taccum = trend.
// ---------------------------------------------------------------------------
__global__ void k_decomp(const float* __restrict__ x, float* __restrict__ seasonal,
                         float* __restrict__ taccum, int L, int C, int mode, long total) {
  long gid = (long)blockIdx.x * blockDim.x + threadIdx.x;
  if (gid >= total) return;
  int c = (int)(gid % C);
  int l = (int)((gid / C) % L);
  long b = gid / ((long)C * L);
  const float* xb = x + b * (long)L * C;
  float s = 0.f;
#pragma unroll 5
  for (int j = -12; j <= 12; ++j) {
    int ll = l + j;
    ll = ll < 0 ? 0 : (ll >= L ? L - 1 : ll);
    s += xb[(long)ll * C + c];
  }
  float tr = s * (1.0f / (float)MAK);
  seasonal[gid] = x[gid] - tr;
  if (mode == 1) taccum[gid] += tr;
  else if (mode == 2) taccum[gid] = tr;
}

// ---------------------------------------------------------------------------
// Kernel 4: per-(b,c) mean over L (for trend_init tail)
// ---------------------------------------------------------------------------
__global__ void k_mean_bc(const float* __restrict__ x, float* __restrict__ m, int L, int C, int total) {
  int gid = blockIdx.x * blockDim.x + threadIdx.x;
  if (gid >= total) return;
  int c = gid % C, b = gid / C;
  float s = 0.f;
  for (int l = 0; l < L; ++l) s += x[((long)b * L + l) * C + c];
  m[gid] = s / (float)L;
}

// ---------------------------------------------------------------------------
// Kernel 5: build trend_init (into trend accumulator) and seasonal_init
// ---------------------------------------------------------------------------
__global__ void k_build_inits(const float* __restrict__ seas, const float* __restrict__ tr,
                              const float* __restrict__ meanbc,
                              float* __restrict__ trend_out, float* __restrict__ seas_out,
                              int L, int C, long total) {
  long gid = (long)blockIdx.x * blockDim.x + threadIdx.x;
  if (gid >= total) return;
  int c = (int)(gid % C);
  int l = (int)((gid / C) % L);
  long b = gid / ((long)C * L);
  if (l < L - PRED) {
    long src = (b * L + (l + PRED)) * (long)C + c;
    trend_out[gid] = tr[src];
    seas_out[gid]  = seas[src];
  } else {
    trend_out[gid] = meanbc[b * C + c];
    seas_out[gid]  = 0.f;
  }
}

// ---------------------------------------------------------------------------
// Kernel 6: data embedding = circular conv3(x) + mark @ timeW + pos_embed
// ---------------------------------------------------------------------------
__global__ void k_embed(const float* __restrict__ x, const float* __restrict__ mark,
                        const float* __restrict__ valW, const float* __restrict__ timeW,
                        float* __restrict__ out, int L, int Cin, long total) {
  long gid = (long)blockIdx.x * blockDim.x + threadIdx.x;
  if (gid >= total) return;
  int d = (int)(gid % DMOD);
  int l = (int)((gid / DMOD) % L);
  long b = gid / ((long)DMOD * L);
  float s = 0.f;
#pragma unroll
  for (int t = 0; t < 3; ++t) {
    int ll = l - 1 + t;
    ll = (ll + L) % L;
    const float* xr = x + (b * L + ll) * (long)Cin;
    const float* wv = valW + (long)t * Cin * DMOD + d;
    for (int c = 0; c < Cin; ++c) s += xr[c] * wv[(long)c * DMOD];
  }
  const float* mr = mark + (b * L + l) * (long)CMK;
#pragma unroll
  for (int m = 0; m < CMK; ++m) s += mr[m] * timeW[(long)m * DMOD + d];
  float freq = expf(-(float)(d & ~1) * (9.210340371976184f / (float)DMOD));
  s += (d & 1) ? cosf((float)l * freq) : sinf((float)l * freq);
  out[gid] = s;
}

// ---------------------------------------------------------------------------
// Kernel 7: autocorrelation scores (time-domain equivalent of rfft path)
// corr[b,lag] = mean over channels of sum_t q[b,t,:] * k[b,(t+lag)%L,:]
// ---------------------------------------------------------------------------
__global__ void k_corr(const float* __restrict__ q, const float* __restrict__ kk,
                       float* __restrict__ corr, int L, int D) {
  int lag = blockIdx.x;
  int b   = blockIdx.y;
  __shared__ float red[256];
  const float* qb = q  + (long)b * L * D;
  const float* kb = kk + (long)b * L * D;
  float s = 0.f;
  for (int i = threadIdx.x; i < L * D; i += 256) {
    int t = i / D, d = i - t * D;
    int t2 = t + lag; if (t2 >= L) t2 -= L;
    s += qb[(long)t * D + d] * kb[(long)t2 * D + d];
  }
  red[threadIdx.x] = s;
  __syncthreads();
  for (int off = 128; off > 0; off >>= 1) {
    if (threadIdx.x < off) red[threadIdx.x] += red[threadIdx.x + off];
    __syncthreads();
  }
  if (threadIdx.x == 0) corr[(long)b * L + lag] = red[0] / (float)D;
}

// ---------------------------------------------------------------------------
// Kernel 8: top-k + softmax over delays (k = 6 = int(log 512)), per batch
// ---------------------------------------------------------------------------
__global__ void k_topk(const float* __restrict__ corr, float* __restrict__ w,
                       int* __restrict__ delay, int L) {
  int b = blockIdx.x;
  if (threadIdx.x != 0) return;
  float vals[TOPK]; int idx[TOPK];
  for (int t = 0; t < TOPK; ++t) {
    float best = -3.4e38f; int bi = 0;
    for (int l = 0; l < L; ++l) {
      float c = corr[(long)b * L + l];
      bool used = false;
      for (int j = 0; j < t; ++j) if (idx[j] == l) used = true;
      if (!used && c > best) { best = c; bi = l; }
    }
    vals[t] = best; idx[t] = bi;
  }
  float mx = vals[0];
  for (int j = 1; j < TOPK; ++j) mx = fmaxf(mx, vals[j]);
  float ssum = 0.f;
  for (int j = 0; j < TOPK; ++j) { vals[j] = expf(vals[j] - mx); ssum += vals[j]; }
  for (int j = 0; j < TOPK; ++j) { w[b * TOPK + j] = vals[j] / ssum; delay[b * TOPK + j] = idx[j]; }
}

// ---------------------------------------------------------------------------
// Kernel 9: delayed aggregation  out[b,l,d] = sum_k w[b,k] * v[b,(l+delay)%L,d]
// ---------------------------------------------------------------------------
__global__ void k_agg(const float* __restrict__ v, const float* __restrict__ w,
                      const int* __restrict__ delay, float* __restrict__ out,
                      int L, int D, long total) {
  long gid = (long)blockIdx.x * blockDim.x + threadIdx.x;
  if (gid >= total) return;
  int d = (int)(gid % D);
  int l = (int)((gid / D) % L);
  long b = gid / ((long)D * L);
  float s = 0.f;
#pragma unroll
  for (int t = 0; t < TOPK; ++t) {
    int ll = l + delay[b * TOPK + t];
    if (ll >= L) ll -= L;
    s += w[b * TOPK + t] * v[(b * L + ll) * (long)D + d];
  }
  out[gid] = s;
}

// ---------------------------------------------------------------------------
// Kernel 10: elementwise helpers
// ---------------------------------------------------------------------------
__global__ void k_add_inplace(float* __restrict__ a, const float* __restrict__ b, long n) {
  long i = (long)blockIdx.x * blockDim.x + threadIdx.x;
  if (i < n) a[i] += b[i];
}
__global__ void k_zero(float* __restrict__ a, long n) {
  long i = (long)blockIdx.x * blockDim.x + threadIdx.x;
  if (i < n) a[i] = 0.f;
}

// ---------------------------------------------------------------------------
// Kernel 11: layernorm over D (one block per (b,l))
// ---------------------------------------------------------------------------
__global__ __launch_bounds__(256) void k_ln(const float* __restrict__ x,
                                            const float* __restrict__ g,
                                            const float* __restrict__ bta,
                                            float* __restrict__ out, int D) {
  long bl = blockIdx.x;
  const float* xr = x + bl * D;
  __shared__ float red[256];
  float s = 0.f;
  for (int d = threadIdx.x; d < D; d += 256) s += xr[d];
  red[threadIdx.x] = s; __syncthreads();
  for (int off = 128; off > 0; off >>= 1) {
    if (threadIdx.x < off) red[threadIdx.x] += red[threadIdx.x + off];
    __syncthreads();
  }
  float mu = red[0] / (float)D;
  __syncthreads();
  float s2 = 0.f;
  for (int d = threadIdx.x; d < D; d += 256) { float t = xr[d] - mu; s2 += t * t; }
  red[threadIdx.x] = s2; __syncthreads();
  for (int off = 128; off > 0; off >>= 1) {
    if (threadIdx.x < off) red[threadIdx.x] += red[threadIdx.x + off];
    __syncthreads();
  }
  float inv = rsqrtf(red[0] / (float)D + 1e-5f);
  for (int d = threadIdx.x; d < D; d += 256)
    out[bl * D + d] = (xr[d] - mu) * inv * g[d] + bta[d];
}

// Kernel 12: subtract per-(b,d) mean over L (seasonal_ln second stage)
__global__ void k_subcolmean(float* __restrict__ x, int L, int D, int total) {
  int gid = blockIdx.x * blockDim.x + threadIdx.x;
  if (gid >= total) return;
  int d = gid % D, b = gid / D;
  float s = 0.f;
  for (int l = 0; l < L; ++l) s += x[((long)b * L + l) * D + d];
  s /= (float)L;
  for (int l = 0; l < L; ++l) x[((long)b * L + l) * D + d] -= s;
}

// ---------------------------------------------------------------------------
// Kernel 13: trend residual = circular conv3(t1+t2+t3) @ trend_W, += trend
// ---------------------------------------------------------------------------
__global__ void k_trendconv(const float* __restrict__ tsum, const float* __restrict__ W,
                            float* __restrict__ trend, int L, long total) {
  long gid = (long)blockIdx.x * blockDim.x + threadIdx.x;
  if (gid >= total) return;
  int c = (int)(gid % COUT);
  int l = (int)((gid / COUT) % L);
  long b = gid / ((long)COUT * L);
  float s = 0.f;
#pragma unroll
  for (int t = 0; t < 3; ++t) {
    int ll = l - 1 + t;
    ll = (ll + L) % L;
    const float* tr = tsum + (b * L + ll) * (long)DMOD;
    const float* wv = W + (long)t * DMOD * COUT + c;
    for (int d = 0; d < DMOD; ++d) s += tr[d] * wv[(long)d * COUT];
  }
  trend[gid] += s;
}

// ---------------------------------------------------------------------------
// Kernel 14: final projection + trend add, keep last PRED steps
// ---------------------------------------------------------------------------
__global__ void k_final(const float* __restrict__ dec, const float* __restrict__ projW,
                        const float* __restrict__ projb, const float* __restrict__ trend,
                        float* __restrict__ out, int L, long total) {
  long gid = (long)blockIdx.x * blockDim.x + threadIdx.x;
  if (gid >= total) return;
  int c  = (int)(gid % COUT);
  int lp = (int)((gid / COUT) % PRED);
  long b = gid / ((long)COUT * PRED);
  int l = lp + (L - PRED);
  const float* dr = dec + (b * L + l) * (long)DMOD;
  float s = projb[c];
  for (int d = 0; d < DMOD; ++d) s += dr[d] * projW[(long)d * COUT + c];
  out[gid] = s + trend[(b * L + l) * (long)COUT + c];
}

// ---------------------------------------------------------------------------
// Host orchestration
// ---------------------------------------------------------------------------
extern "C" void kernel_launch(void* const* d_in, const int* in_sizes, int n_in,
                              void* d_out, int out_size, void* d_ws, size_t ws_size,
                              hipStream_t stream) {
  (void)in_sizes; (void)n_in; (void)out_size; (void)ws_size;

  const float* x_enc      = (const float*)d_in[0];
  const float* x_mark_enc = (const float*)d_in[1];
  const float* x_mark_dec = (const float*)d_in[3];
  const float* enc_val_W  = (const float*)d_in[4];
  const float* enc_time_W = (const float*)d_in[5];
  const float* enc_Wq     = (const float*)d_in[6];
  const float* enc_Wk     = (const float*)d_in[7];
  const float* enc_Wv     = (const float*)d_in[8];
  const float* enc_Wo     = (const float*)d_in[9];
  const float* enc_W1     = (const float*)d_in[10];
  const float* enc_W2     = (const float*)d_in[11];
  const float* enc_ng     = (const float*)d_in[12];
  const float* enc_nb     = (const float*)d_in[13];
  const float* dec_val_W  = (const float*)d_in[14];
  const float* dec_time_W = (const float*)d_in[15];
  const float* dec_sWq    = (const float*)d_in[16];
  const float* dec_sWk    = (const float*)d_in[17];
  const float* dec_sWv    = (const float*)d_in[18];
  const float* dec_sWo    = (const float*)d_in[19];
  const float* dec_cWq    = (const float*)d_in[20];
  const float* dec_cWk    = (const float*)d_in[21];
  const float* dec_cWv    = (const float*)d_in[22];
  const float* dec_cWo    = (const float*)d_in[23];
  const float* dec_W1     = (const float*)d_in[24];
  const float* dec_W2     = (const float*)d_in[25];
  const float* dec_trend_W= (const float*)d_in[26];
  const float* dec_ng     = (const float*)d_in[27];
  const float* dec_nb     = (const float*)d_in[28];
  const float* proj_W     = (const float*)d_in[29];
  const float* proj_b     = (const float*)d_in[30];

  const long BLD  = (long)BN * LSEQ * DMOD;   // 8.39M
  const long BLDF = (long)BN * LSEQ * DFF;    // 33.6M
  const long BLC  = (long)BN * LSEQ * CIN;    // 344k
  const int  MBL  = BN * LSEQ;                // 16384 rows

  char* p = (char*)d_ws;
  auto alloc = [&](size_t bytes) -> void* {
    void* r = (void*)p;
    p += (bytes + 255) & ~(size_t)255;
    return r;
  };

  // bf16 transposed weight mirrors [N,K]
  __bf16* wEQ = (__bf16*)alloc((size_t)2 * DMOD * DMOD * 2);
  __bf16* wEK = (__bf16*)alloc((size_t)2 * DMOD * DMOD * 2);
  __bf16* wEV = (__bf16*)alloc((size_t)2 * DMOD * DMOD * 2);
  __bf16* wEO = (__bf16*)alloc((size_t)2 * DMOD * DMOD * 2);
  __bf16* wE1 = (__bf16*)alloc((size_t)2 * DMOD * DFF * 2);
  __bf16* wE2 = (__bf16*)alloc((size_t)2 * DFF * DMOD * 2);
  __bf16* wSQ = (__bf16*)alloc((size_t)DMOD * DMOD * 2);
  __bf16* wSK = (__bf16*)alloc((size_t)DMOD * DMOD * 2);
  __bf16* wSV = (__bf16*)alloc((size_t)DMOD * DMOD * 2);
  __bf16* wSO = (__bf16*)alloc((size_t)DMOD * DMOD * 2);
  __bf16* wCQ = (__bf16*)alloc((size_t)DMOD * DMOD * 2);
  __bf16* wCK = (__bf16*)alloc((size_t)DMOD * DMOD * 2);
  __bf16* wCV = (__bf16*)alloc((size_t)DMOD * DMOD * 2);
  __bf16* wCO = (__bf16*)alloc((size_t)DMOD * DMOD * 2);
  __bf16* wD1 = (__bf16*)alloc((size_t)DMOD * DFF * 2);
  __bf16* wD2 = (__bf16*)alloc((size_t)DFF * DMOD * 2);

  // bf16 activation operand scratch (sized for the largest GEMM A: [MBL,DFF])
  __bf16* abf = (__bf16*)alloc((size_t)BLDF * 2);

  // fp32 activation buffers
  float* xe   = (float*)alloc(BLD * 4);
  float* xd   = (float*)alloc(BLD * 4);
  float* bq   = (float*)alloc(BLD * 4);
  float* bk   = (float*)alloc(BLD * 4);
  float* bv   = (float*)alloc(BLD * 4);
  float* bt   = (float*)alloc(BLD * 4);
  float* bs   = (float*)alloc(BLD * 4);
  float* tsum = (float*)alloc(BLD * 4);
  float* ff   = (float*)alloc(BLDF * 4);

  float* seas_s   = (float*)alloc(BLC * 4);
  float* trend_s  = (float*)alloc(BLC * 4);
  float* mean_bc  = (float*)alloc((size_t)BN * CIN * 4);
  float* trendBuf = (float*)alloc(BLC * 4);
  float* seasInit = (float*)alloc(BLC * 4);
  float* corr     = (float*)alloc((size_t)BN * LSEQ * 4);
  float* wtop     = (float*)alloc((size_t)BN * TOPK * 4);
  int*   dtop     = (int*)alloc((size_t)BN * TOPK * 4);

  auto g1 = [](long n) { return (unsigned)((n + 255) / 256); };
  auto convT = [&](const float* src, __bf16* dst, int K, int N) {
    k_transpose_bf16<<<g1((long)K * N), 256, 0, stream>>>(src, dst, K, N);
  };
  auto convA = [&](const float* src, long n) {
    k_f32_to_bf16<<<g1(n), 256, 0, stream>>>(src, abf, n);
  };
  auto gemm = [&](const __bf16* A, const __bf16* Wt, float* Cc, int M, int N, int K, int act) {
    dim3 g(N / TN, M / TM);
    k_gemm_bf16<<<g, 256, 0, stream>>>(A, Wt, Cc, M, N, K, act);
  };

  // ---- weight transpose-conversion ----
  const long off2 = (long)DMOD * DMOD;
  const long offF = (long)DMOD * DFF;
  for (int i = 0; i < 2; ++i) {
    convT(enc_Wq + i * off2, wEQ + i * off2, DMOD, DMOD);
    convT(enc_Wk + i * off2, wEK + i * off2, DMOD, DMOD);
    convT(enc_Wv + i * off2, wEV + i * off2, DMOD, DMOD);
    convT(enc_Wo + i * off2, wEO + i * off2, DMOD, DMOD);
    convT(enc_W1 + i * offF, wE1 + i * offF, DMOD, DFF);
    convT(enc_W2 + i * offF, wE2 + i * offF, DFF, DMOD);
  }
  convT(dec_sWq, wSQ, DMOD, DMOD);  convT(dec_sWk, wSK, DMOD, DMOD);
  convT(dec_sWv, wSV, DMOD, DMOD);  convT(dec_sWo, wSO, DMOD, DMOD);
  convT(dec_cWq, wCQ, DMOD, DMOD);  convT(dec_cWk, wCK, DMOD, DMOD);
  convT(dec_cWv, wCV, DMOD, DMOD);  convT(dec_cWo, wCO, DMOD, DMOD);
  convT(dec_W1, wD1, DMOD, DFF);    convT(dec_W2, wD2, DFF, DMOD);

  // ---- init decomposition of x_enc ----
  k_decomp<<<g1(BLC), 256, 0, stream>>>(x_enc, seas_s, trend_s, LSEQ, CIN, 2, BLC);
  k_mean_bc<<<g1(BN * CIN), 256, 0, stream>>>(x_enc, mean_bc, LSEQ, CIN, BN * CIN);
  k_build_inits<<<g1(BLC), 256, 0, stream>>>(seas_s, trend_s, mean_bc, trendBuf, seasInit,
                                             LSEQ, CIN, BLC);

  // ---- encoder embedding ----
  k_embed<<<g1(BLD), 256, 0, stream>>>(x_enc, x_mark_enc, enc_val_W, enc_time_W, xe,
                                       LSEQ, CIN, BLD);

  auto attention = [&](const float* qx, const float* kx,
                       const __bf16* Wq, const __bf16* Wk, const __bf16* Wv,
                       const __bf16* Wo, float* xio) {
    convA(qx, BLD);
    gemm(abf, Wq, bq, MBL, DMOD, DMOD, 0);
    if (kx != qx) convA(kx, BLD);
    gemm(abf, Wk, bk, MBL, DMOD, DMOD, 0);
    gemm(abf, Wv, bv, MBL, DMOD, DMOD, 0);
    k_corr<<<dim3(LSEQ, BN), 256, 0, stream>>>(bq, bk, corr, LSEQ, DMOD);
    k_topk<<<BN, 32, 0, stream>>>(corr, wtop, dtop, LSEQ);
    k_agg<<<g1(BLD), 256, 0, stream>>>(bv, wtop, dtop, bt, LSEQ, DMOD, BLD);
    convA(bt, BLD);
    gemm(abf, Wo, bq, MBL, DMOD, DMOD, 0);
    k_add_inplace<<<g1(BLD), 256, 0, stream>>>(xio, bq, BLD);
  };

  // ---- encoder layers ----
  for (int i = 0; i < 2; ++i) {
    attention(xe, xe, wEQ + i * off2, wEK + i * off2, wEV + i * off2, wEO + i * off2, xe);
    k_decomp<<<g1(BLD), 256, 0, stream>>>(xe, bs, nullptr, LSEQ, DMOD, 0, BLD);
    { float* t = xe; xe = bs; bs = t; }
    convA(xe, BLD);
    gemm(abf, wE1 + i * offF, ff, MBL, DFF, DMOD, 1);   // fused GELU
    convA(ff, BLDF);
    gemm(abf, wE2 + i * offF, bq, MBL, DMOD, DFF, 0);
    k_add_inplace<<<g1(BLD), 256, 0, stream>>>(xe, bq, BLD);
    k_decomp<<<g1(BLD), 256, 0, stream>>>(xe, bs, nullptr, LSEQ, DMOD, 0, BLD);
    { float* t = xe; xe = bs; bs = t; }
  }
  // encoder seasonal layernorm
  k_ln<<<MBL, 256, 0, stream>>>(xe, enc_ng, enc_nb, bs, DMOD);
  k_subcolmean<<<g1(BN * DMOD), 256, 0, stream>>>(bs, LSEQ, DMOD, BN * DMOD);
  { float* t = xe; xe = bs; bs = t; }   // xe = encoder output

  // ---- decoder ----
  k_embed<<<g1(BLD), 256, 0, stream>>>(seasInit, x_mark_dec, dec_val_W, dec_time_W, xd,
                                       LSEQ, CIN, BLD);
  k_zero<<<g1(BLD), 256, 0, stream>>>(tsum, BLD);

  // self attention
  attention(xd, xd, wSQ, wSK, wSV, wSO, xd);
  k_decomp<<<g1(BLD), 256, 0, stream>>>(xd, bs, tsum, LSEQ, DMOD, 1, BLD);
  { float* t = xd; xd = bs; bs = t; }
  // cross attention (enc output as K/V source)
  attention(xd, xe, wCQ, wCK, wCV, wCO, xd);
  k_decomp<<<g1(BLD), 256, 0, stream>>>(xd, bs, tsum, LSEQ, DMOD, 1, BLD);
  { float* t = xd; xd = bs; bs = t; }
  // FFN
  convA(xd, BLD);
  gemm(abf, wD1, ff, MBL, DFF, DMOD, 1);
  convA(ff, BLDF);
  gemm(abf, wD2, bq, MBL, DMOD, DFF, 0);
  k_add_inplace<<<g1(BLD), 256, 0, stream>>>(xd, bq, BLD);
  k_decomp<<<g1(BLD), 256, 0, stream>>>(xd, bs, tsum, LSEQ, DMOD, 1, BLD);
  { float* t = xd; xd = bs; bs = t; }
  // residual trend
  k_trendconv<<<g1(BLC), 256, 0, stream>>>(tsum, dec_trend_W, trendBuf, LSEQ, BLC);

  // decoder seasonal layernorm
  k_ln<<<MBL, 256, 0, stream>>>(xd, dec_ng, dec_nb, bs, DMOD);
  k_subcolmean<<<g1(BN * DMOD), 256, 0, stream>>>(bs, LSEQ, DMOD, BN * DMOD);
  { float* t = xd; xd = bs; bs = t; }

  // final projection + trend, last PRED steps
  const long outN = (long)BN * PRED * COUT;
  k_final<<<g1(outN), 256, 0, stream>>>(xd, proj_W, proj_b, trendBuf, (float*)d_out,
                                        LSEQ, outN);
}
